// LstmDecoder_25134148617009
// MI455X (gfx1250) — matrix operand order repaired and checked by reading, not compiled, not run on hardware
//
#include <hip/hip_runtime.h>

#define B_      64
#define T_      127
#define S_      128
#define ENC_    2048
#define EMB_    512
#define HID_    512
#define VOCAB_  10000
#define G4_     2048
#define NBLK_LSTM 64

typedef __attribute__((ext_vector_type(16))) __bf16 v16bf;
typedef __attribute__((ext_vector_type(8)))  float  v8f;
typedef __attribute__((__vector_size__(4 * sizeof(int)))) int vsi4;

#if __has_builtin(__builtin_amdgcn_global_load_async_to_lds_b128) && \
    __has_builtin(__builtin_amdgcn_s_wait_asynccnt)
#define ASYNC_LDS_OK 1
#endif

#ifdef ASYNC_LDS_OK
__device__ __forceinline__ void async_copy_b128(const void* g, void* l) {
  __builtin_amdgcn_global_load_async_to_lds_b128(
      (__attribute__((address_space(1))) vsi4*)(void*)g,
      (__attribute__((address_space(3))) vsi4*)l, 0, 0);
}
#endif

__device__ __forceinline__ __bf16 f2bf(float f) {
  union { float f; unsigned u; } a; a.f = f;
  unsigned r = a.u + 0x7FFFu + ((a.u >> 16) & 1u);   // round-to-nearest-even
  union { unsigned short s; __bf16 b; } o; o.s = (unsigned short)(r >> 16);
  return o.b;
}

__device__ __forceinline__ v16bf ld2(const __bf16* p0, const __bf16* p1) {
  union { uint4 q[2]; v16bf v; } u;
  u.q[0] = *(const uint4*)p0;
  u.q[1] = *(const uint4*)p1;
  return u.v;
}

__device__ __forceinline__ v8f zero8() {
  v8f a;
#pragma unroll
  for (int i = 0; i < 8; ++i) a[i] = 0.f;
  return a;
}

__device__ __forceinline__ v8f wmma_bf16(v16bf a, v16bf b, v8f c) {
  return __builtin_amdgcn_wmma_f32_16x16x32_bf16(false, a, false, b, (short)0, c,
                                                 false, false);
}

__device__ __forceinline__ float sigm(float x) { return 1.f / (1.f + __expf(-x)); }

// Device-scope barrier across the persistent LSTM grid.
__device__ __forceinline__ void gbar(unsigned* bar, unsigned target) {
  __threadfence();
  __syncthreads();
  if (threadIdx.x == 0) {
    __hip_atomic_fetch_add(bar, 1u, __ATOMIC_RELEASE, __HIP_MEMORY_SCOPE_AGENT);
    while (__hip_atomic_load(bar, __ATOMIC_ACQUIRE, __HIP_MEMORY_SCOPE_AGENT) < target)
      __builtin_amdgcn_s_sleep(1);
  }
  __syncthreads();
}

// ---------------- K1: fc1 (fp32, tiny) ----------------
__global__ __launch_bounds__(256) void fc1_kernel(const float* __restrict__ x,
                                                  const float* __restrict__ w,
                                                  const float* __restrict__ bias,
                                                  float* __restrict__ feat) {
  int idx = blockIdx.x * 256 + threadIdx.x;       // 64*512
  int b = idx >> 9, e = idx & 511;
  const float* xr = x + (size_t)b * ENC_;
  const float* wr = w + (size_t)e * ENC_;
  float acc = bias[e];
  for (int k = 0; k < ENC_; k += 4) {
    float4 xv = *(const float4*)(xr + k);
    float4 wv = *(const float4*)(wr + k);
    acc += xv.x * wv.x + xv.y * wv.y + xv.z * wv.z + xv.w * wv.w;
  }
  feat[idx] = acc;
}

// ---------------- K2: batchnorm (batch stats) -> seq[:,0,:] bf16 ----------------
__global__ void bn_kernel(const float* __restrict__ feat,
                          const float* __restrict__ gamma,
                          const float* __restrict__ beta,
                          __bf16* __restrict__ seq) {
  int c = blockIdx.x, b = threadIdx.x;            // 512 blocks x 64 threads
  float v = feat[(size_t)b * EMB_ + c];
  __shared__ float s1[64], s2[64];
  s1[b] = v; s2[b] = v * v; __syncthreads();
  for (int off = 32; off > 0; off >>= 1) {
    if (b < off) { s1[b] += s1[b + off]; s2[b] += s2[b + off]; }
    __syncthreads();
  }
  float mu = s1[0] * (1.f / 64.f);
  float var = s2[0] * (1.f / 64.f) - mu * mu;
  float sc = rsqrtf(var + 1e-5f) * gamma[c];
  seq[(size_t)b * S_ * EMB_ + c] = f2bf((v - mu) * sc + beta[c]);
}

// ---------------- K3: embedding gather -> seq[:,1:,:] bf16 ----------------
__global__ void embed_kernel(const int* __restrict__ y,
                             const float* __restrict__ emb,
                             __bf16* __restrict__ seq) {
  int i = blockIdx.x * 256 + threadIdx.x;         // 64*127*512
  int d = i & 511, r = i >> 9;
  int t = r % T_, b = r / T_;
  int tok = y[b * T_ + t];
  seq[(size_t)b * S_ * EMB_ + (size_t)(t + 1) * EMB_ + d] =
      f2bf(emb[(size_t)tok * EMB_ + d]);
}

// ---------------- K4: weight conversion (bf16) + bias fold ----------------
__global__ void conv_kernel(const float* __restrict__ w_ih, const float* __restrict__ w_hh,
                            const float* __restrict__ b_ih, const float* __restrict__ b_hh,
                            const float* __restrict__ fc2w,
                            __bf16* __restrict__ wcat, float* __restrict__ bias,
                            __bf16* __restrict__ w2) {
  const int NW = G4_ * 1024, NW2 = VOCAB_ * HID_;
  int i = blockIdx.x * 256 + threadIdx.x;
  if (i < NW) {
    int n = i >> 10, k = i & 1023;                // [xh ; hh] concat along K
    float v = (k < 512) ? w_ih[(size_t)n * 512 + k] : w_hh[(size_t)n * 512 + (k - 512)];
    wcat[i] = f2bf(v);
  } else if (i < NW + NW2) {
    int j = i - NW;
    w2[j] = f2bf(fc2w[j]);
  } else if (i < NW + NW2 + G4_) {
    int j = i - NW - NW2;
    bias[j] = b_ih[j] + b_hh[j];
  }
}

// ---------------- K5: persistent LSTM, weights LDS-resident ----------------
__global__ __launch_bounds__(256) void lstm_kernel(
    const __bf16* __restrict__ seq, const __bf16* __restrict__ wcat,
    const float* __restrict__ bias, const int* __restrict__ lengths,
    __bf16* __restrict__ h_bf, float* __restrict__ c_buf,
    float* __restrict__ gates, __bf16* __restrict__ hs, __bf16* __restrict__ hsT,
    unsigned* __restrict__ bar) {
  __shared__ __bf16 lw[32 * 1024];                // 64 KB: this WG's 32 gate cols x K=1024
  const int tid = threadIdx.x, bx = blockIdx.x;
#ifdef ASYNC_LDS_OK
  {
    const char* src = (const char*)(wcat + (size_t)bx * 32 * 1024);
    char* dst = (char*)lw;
    for (int i = tid; i < 32 * 1024 * 2 / 16; i += 256)
      async_copy_b128(src + (size_t)i * 16, dst + (size_t)i * 16);
    __builtin_amdgcn_s_wait_asynccnt(0);
  }
#else
  {
    const uint4* src = (const uint4*)(wcat + (size_t)bx * 32 * 1024);
    uint4* dst = (uint4*)lw;
    for (int i = tid; i < 32 * 1024 * 2 / 16; i += 256) dst[i] = src[i];
  }
#endif
  __syncthreads();

  const int wave = tid >> 5, lane = tid & 31;
  const int mi = wave & 3, ni = wave >> 2;        // 4 M-tiles x 2 N-tiles
  const int lanelo = lane & 15, lanehi = lane >> 4;
  unsigned phase = 0;

  for (int t = 0; t < S_; ++t) {
    // gates[64 x 32slice] = [x_t ; h] @ Wslice^T   (M=64,N=32,K=1024)
    v8f acc = zero8();
    const int arow = mi * 16 + lanelo;
    for (int kk = 0; kk < 32; ++kk) {
      const int k0 = kk * 32;
      const int ka = k0 + lanehi * 8;
      const __bf16* ap = (k0 < 512)
          ? seq + (size_t)arow * S_ * EMB_ + (size_t)t * EMB_ + ka
          : h_bf + (size_t)arow * HID_ + (ka - 512);
      v16bf a = ld2(ap, ap + 16);
      const __bf16* bp = lw + (ni * 16 + lanelo) * 1024 + k0 + lanehi * 8;
      v16bf b = ld2(bp, bp + 16);
      acc = wmma_bf16(a, b, acc);
    }
    const int ng = bx * 32 + ni * 16 + lanelo;
#pragma unroll
    for (int r = 0; r < 8; ++r) {
      int m = mi * 16 + r + lanehi * 8;
      gates[(size_t)m * G4_ + ng] = acc[r];
    }
    gbar(bar, (++phase) * NBLK_LSTM);

    // elementwise: this WG owns hidden units [bx*8, bx*8+8), all 64 batches
#pragma unroll
    for (int it = 0; it < 2; ++it) {
      int e = it * 256 + tid;                     // 0..511
      int b = e >> 3, j = bx * 8 + (e & 7);
      float gi = gates[(size_t)b * G4_ + j]            + bias[j];
      float gf = gates[(size_t)b * G4_ + 512 + j]      + bias[512 + j];
      float gg = gates[(size_t)b * G4_ + 1024 + j]     + bias[1024 + j];
      float go = gates[(size_t)b * G4_ + 1536 + j]     + bias[1536 + j];
      float i_ = sigm(gi), f_ = sigm(gf), g_ = tanhf(gg), o_ = sigm(go);
      float c = f_ * c_buf[(size_t)b * HID_ + j] + i_ * g_;
      c_buf[(size_t)b * HID_ + j] = c;
      float h = o_ * tanhf(c);
      h_bf[(size_t)b * HID_ + j] = f2bf(h);
      float hm = (t < lengths[b]) ? h : 0.f;      // pad_packed_sequence mask
      hs [(size_t)b * S_ * HID_ + (size_t)t * HID_ + j] = f2bf(hm);
      hsT[(size_t)b * HID_ * S_ + (size_t)j * S_ + t]   = f2bf(hm);
    }
    gbar(bar, (++phase) * NBLK_LSTM);
  }
}

// ---------------- K6a: scores = Hs @ Hs^T per batch (WMMA, A reused 8x) ----------------
__global__ __launch_bounds__(256) void scores_kernel(const __bf16* __restrict__ hs,
                                                     float* __restrict__ scores) {
  int b = blockIdx.x;
  int wave = threadIdx.x >> 5, lane = threadIdx.x & 31;
  int lanelo = lane & 15, lanehi = lane >> 4;
  const __bf16* H = hs + (size_t)b * S_ * HID_;
  const int mi = wave;                            // 8 M-tiles of t
  v8f acc[8];
#pragma unroll
  for (int n = 0; n < 8; ++n) acc[n] = zero8();
  for (int kk = 0; kk < 16; ++kk) {
    int k0 = kk * 32;
    const __bf16* ap = H + (size_t)(mi * 16 + lanelo) * HID_ + k0 + lanehi * 8;
    v16bf a = ld2(ap, ap + 16);
#pragma unroll
    for (int n = 0; n < 8; ++n) {
      const __bf16* bp = H + (size_t)(n * 16 + lanelo) * HID_ + k0 + lanehi * 8;
      v16bf bb = ld2(bp, bp + 16);
      acc[n] = wmma_bf16(a, bb, acc[n]);
    }
  }
#pragma unroll
  for (int n = 0; n < 8; ++n)
#pragma unroll
    for (int r = 0; r < 8; ++r) {
      int tr = mi * 16 + r + lanehi * 8, sc = n * 16 + lanelo;
      scores[(size_t)b * S_ * S_ + (size_t)tr * S_ + sc] = acc[n][r];
    }
}

// ---------------- K6b: causal-masked softmax -> alpha bf16 ----------------
__global__ void softmax_kernel(const float* __restrict__ scores,
                               __bf16* __restrict__ alpha) {
  int row = blockIdx.x;                           // b*S + t
  int t = row & (S_ - 1), s = threadIdx.x;
  float v = (s < t) ? scores[(size_t)row * S_ + s] : 0.f;  // zeroed keys -> score 0
  __shared__ float red[S_];
  red[s] = v; __syncthreads();
  for (int off = 64; off > 0; off >>= 1) {
    if (s < off) red[s] = fmaxf(red[s], red[s + off]);
    __syncthreads();
  }
  float mx = red[0]; __syncthreads();
  float e = __expf(v - mx);
  red[s] = e; __syncthreads();
  for (int off = 64; off > 0; off >>= 1) {
    if (s < off) red[s] += red[s + off];
    __syncthreads();
  }
  float a = e / red[0];
  alpha[(size_t)row * S_ + s] = f2bf((s < t) ? a : 0.f);   // alpha * cmask
}

// ---------------- K7: attn = alpha @ Hs per batch (WMMA, full-K A cached) ----------------
__global__ __launch_bounds__(256) void attn_kernel(const __bf16* __restrict__ alpha,
                                                   const __bf16* __restrict__ hsT,
                                                   __bf16* __restrict__ attn) {
  int b = blockIdx.x;
  int wave = threadIdx.x >> 5, lane = threadIdx.x & 31;
  int lanelo = lane & 15, lanehi = lane >> 4;
  const __bf16* A  = alpha + (size_t)b * S_ * S_;
  const __bf16* Bt = hsT   + (size_t)b * HID_ * S_;
  const int mi = wave;
  v16bf a[4];                                     // whole K=128 of this M-tile
#pragma unroll
  for (int kk = 0; kk < 4; ++kk) {
    const __bf16* ap = A + (size_t)(mi * 16 + lanelo) * S_ + kk * 32 + lanehi * 8;
    a[kk] = ld2(ap, ap + 16);
  }
  for (int ni = 0; ni < 32; ++ni) {
    v8f acc = zero8();
#pragma unroll
    for (int kk = 0; kk < 4; ++kk) {
      const __bf16* bp = Bt + (size_t)(ni * 16 + lanelo) * S_ + kk * 32 + lanehi * 8;
      v16bf bb = ld2(bp, bp + 16);
      acc = wmma_bf16(a[kk], bb, acc);
    }
#pragma unroll
    for (int r = 0; r < 8; ++r) {
      int tr = mi * 16 + r + lanehi * 8, d = ni * 16 + lanelo;
      attn[(size_t)b * S_ * HID_ + (size_t)tr * HID_ + d] = f2bf(acc[r]);
    }
  }
}

// ---------------- K8: fc2 (dominant GEMM, 8192x10000x512) ----------------
// 16-col weight panel in LDS (async-staged), 4 M-tiles per wave -> B reused 4x.
__global__ __launch_bounds__(256) void fc2_kernel(const __bf16* __restrict__ attn,
                                                  const __bf16* __restrict__ w2,
                                                  const float* __restrict__ bias,
                                                  float* __restrict__ out) {
  __shared__ __bf16 lb[16 * 512];                 // 16 KB weight panel
  const int v0 = blockIdx.x * 16;
#ifdef ASYNC_LDS_OK
  {
    const char* src = (const char*)(w2 + (size_t)v0 * HID_);
    char* dst = (char*)lb;
    for (int i = threadIdx.x; i < 16 * 512 * 2 / 16; i += 256)
      async_copy_b128(src + (size_t)i * 16, dst + (size_t)i * 16);
    __builtin_amdgcn_s_wait_asynccnt(0);
  }
#else
  {
    const uint4* src = (const uint4*)(w2 + (size_t)v0 * HID_);
    uint4* dst = (uint4*)lb;
    for (int i = threadIdx.x; i < 16 * 512 * 2 / 16; i += 256) dst[i] = src[i];
  }
#endif
  __syncthreads();
  int wave = threadIdx.x >> 5, lane = threadIdx.x & 31;
  int lanelo = lane & 15, lanehi = lane >> 4;
  const int mt0 = (blockIdx.y * 8 + wave) * 4;    // 4 consecutive M-tiles per wave
  v8f acc[4];
#pragma unroll
  for (int u = 0; u < 4; ++u) acc[u] = zero8();
  for (int kk = 0; kk < 16; ++kk) {
    int k0 = kk * 32;
    const __bf16* bp = lb + lanelo * 512 + k0 + lanehi * 8;
    v16bf bb = ld2(bp, bp + 16);
#pragma unroll
    for (int u = 0; u < 4; ++u) {
      const __bf16* ap =
          attn + (size_t)((mt0 + u) * 16 + lanelo) * HID_ + k0 + lanehi * 8;
      v16bf a = ld2(ap, ap + 16);
      acc[u] = wmma_bf16(a, bb, acc[u]);
    }
  }
  const int vcol = v0 + lanelo;
  const float bv = bias[vcol];
#pragma unroll
  for (int u = 0; u < 4; ++u)
#pragma unroll
    for (int r = 0; r < 8; ++r) {
      int m = (mt0 + u) * 16 + r + lanehi * 8;
      out[(size_t)m * VOCAB_ + vcol] = acc[u][r] + bv;
    }
}

// ---------------- host launcher ----------------
extern "C" void kernel_launch(void* const* d_in, const int* in_sizes, int n_in,
                              void* d_out, int out_size, void* d_ws, size_t ws_size,
                              hipStream_t stream) {
  (void)in_sizes; (void)n_in; (void)out_size; (void)ws_size;
  const float* x      = (const float*)d_in[0];
  const int*   y      = (const int*)  d_in[1];
  const int*   len    = (const int*)  d_in[2];
  const float* fc1w   = (const float*)d_in[3];
  const float* fc1b   = (const float*)d_in[4];
  const float* bng    = (const float*)d_in[5];
  const float* bnb    = (const float*)d_in[6];
  const float* emb    = (const float*)d_in[7];
  const float* w_ih   = (const float*)d_in[8];
  const float* w_hh   = (const float*)d_in[9];
  const float* b_ih   = (const float*)d_in[10];
  const float* b_hh   = (const float*)d_in[11];
  const float* fc2w   = (const float*)d_in[12];
  const float* fc2b   = (const float*)d_in[13];
  float* out = (float*)d_out;
  char*  ws  = (char*)d_ws;

  size_t o = 0;
  float*  feat   = (float*) (ws + o); o += (size_t)B_ * EMB_ * 4;            // 128K
  __bf16* seq    = (__bf16*)(ws + o); o += (size_t)B_ * S_ * EMB_ * 2;       // 8M
  __bf16* wcat   = (__bf16*)(ws + o); o += (size_t)G4_ * 1024 * 2;           // 4M
  float*  bias   = (float*) (ws + o); o += (size_t)G4_ * 4;                  // 8K
  __bf16* h_bf   = (__bf16*)(ws + o); o += (size_t)B_ * HID_ * 2;
  float*  c_buf  = (float*) (ws + o); o += (size_t)B_ * HID_ * 4;
  float*  gates  = (float*) (ws + o); o += (size_t)B_ * G4_ * 4;             // 512K
  __bf16* hs     = (__bf16*)(ws + o); o += (size_t)B_ * S_ * HID_ * 2;       // 8M
  __bf16* hsT    = (__bf16*)(ws + o); o += (size_t)B_ * HID_ * S_ * 2;       // 8M
  float*  scores = (float*) (ws + o); o += (size_t)B_ * S_ * S_ * 4;         // 4M
  __bf16* alpha  = (__bf16*)(ws + o); o += (size_t)B_ * S_ * S_ * 2;         // 2M
  __bf16* attn   = (__bf16*)(ws + o); o += (size_t)B_ * S_ * HID_ * 2;       // 8M
  __bf16* w2     = (__bf16*)(ws + o); o += (size_t)VOCAB_ * HID_ * 2;        // 10M
  unsigned* bar  = (unsigned*)(ws + o); o += 256;

  // reset recurrent state + barrier counter (graph-capturable async memsets)
  (void)hipMemsetAsync(bar, 0, 256, stream);
  (void)hipMemsetAsync(h_bf, 0, (size_t)B_ * HID_ * 2, stream);
  (void)hipMemsetAsync(c_buf, 0, (size_t)B_ * HID_ * 4, stream);

  fc1_kernel<<<(B_ * EMB_) / 256, 256, 0, stream>>>(x, fc1w, fc1b, feat);
  bn_kernel<<<EMB_, 64, 0, stream>>>(feat, bng, bnb, seq);
  embed_kernel<<<(B_ * T_ * EMB_) / 256, 256, 0, stream>>>(y, emb, seq);
  {
    int total = G4_ * 1024 + VOCAB_ * HID_ + G4_;
    conv_kernel<<<(total + 255) / 256, 256, 0, stream>>>(w_ih, w_hh, b_ih, b_hh,
                                                         fc2w, wcat, bias, w2);
  }
  lstm_kernel<<<NBLK_LSTM, 256, 0, stream>>>(seq, wcat, bias, len, h_bf, c_buf,
                                             gates, hs, hsT, bar);
  scores_kernel<<<B_, 256, 0, stream>>>(hs, scores);
  softmax_kernel<<<B_ * S_, S_, 0, stream>>>(scores, alpha);
  attn_kernel<<<B_, 256, 0, stream>>>(alpha, hsT, attn);
  dim3 g8(VOCAB_ / 16, B_ * S_ / (8 * 16 * 4));   // (625, 16)
  fc2_kernel<<<g8, 256, 0, stream>>>(attn, w2, fc2b, out);
}
